// SelfAttention_54485955117520
// MI455X (gfx1250) — compile-verified, matrix operand order checked by reading, not run on hardware
//
#include <hip/hip_runtime.h>
#include <hip/hip_bf16.h>
#include <math.h>

// ---------------------------------------------------------------------------
// Single-head causal self-attention for MI455X (gfx1250, wave32, WMMA).
//   x:[8,2048,1024] f32, Wk/Wq/Wv:[128,1024] f32  ->  out:[8,2048,128] f32
// Pass 0: one-time f32->f16 conversion of x and W into workspace.
// Pass 1: q/k/v projections; 64x32 W slices staged to LDS with
//         GLOBAL_LOAD_ASYNC_TO_LDS_B128 (double-buffered, s_wait_asynccnt).
//         Last K-step peeled so the hot loop is single-latch (no phi copies).
// Pass 2: wave-level flash attention, online softmax, WMMA for qk^T and p@v.
// ---------------------------------------------------------------------------

typedef __attribute__((ext_vector_type(16))) _Float16 v16h;
typedef __attribute__((ext_vector_type(8)))  _Float16 v8h;
typedef __attribute__((ext_vector_type(8)))  float    v8f;
typedef __attribute__((ext_vector_type(4)))  float    v4f;
typedef int v4i __attribute__((vector_size(16)));   // matches builtin param

#define BATCH 8
#define TSEQ  2048
#define EMBED 1024
#define HEAD  128
#define BT    (BATCH * TSEQ)
#define KSTEPS (EMBED / 32)

// ---- gfx1250 async global->LDS path (probe via __has_builtin) -------------
#if defined(__has_builtin)
#  if __has_builtin(__builtin_amdgcn_global_load_async_to_lds_b128)
#    define HAVE_ASYNC 1
#  endif
#endif
#ifndef HAVE_ASYNC
#  define HAVE_ASYNC 0
#endif

#if HAVE_ASYNC
typedef __attribute__((address_space(1))) v4i* gptr_b128;
typedef __attribute__((address_space(3))) v4i* lptr_b128;
#  if __has_builtin(__builtin_amdgcn_s_wait_asynccnt)
#    define WAIT_ASYNC(n) __builtin_amdgcn_s_wait_asynccnt(n)
#  else
#    define WAIT_ASYNC(n) asm volatile("s_wait_asynccnt %0" ::"i"(n) : "memory")
#  endif
#else
#  define WAIT_ASYNC(n) ((void)0)
#endif

static __device__ __forceinline__ v8h cvt8h(const float* __restrict__ p) {
  v4f a = *(const v4f*)p;
  v4f b = *(const v4f*)(p + 4);
  v8h r;
  r[0] = (_Float16)a[0]; r[1] = (_Float16)a[1];
  r[2] = (_Float16)a[2]; r[3] = (_Float16)a[3];
  r[4] = (_Float16)b[0]; r[5] = (_Float16)b[1];
  r[6] = (_Float16)b[2]; r[7] = (_Float16)b[3];
  return r;
}

static __device__ __forceinline__ v8h ld8h(const _Float16* __restrict__ p) {
  return *(const v8h*)p;   // 16-byte global_load_b128 / ds_load_b128
}

static __device__ __forceinline__ v16h cat16(v8h lo, v8h hi) {
  return __builtin_shufflevector(lo, hi, 0, 1, 2, 3, 4, 5, 6, 7,
                                         8, 9, 10, 11, 12, 13, 14, 15);
}

static __device__ __forceinline__ v8f wmma16(v16h a, v16h b, v8f c) {
  // D = A(16x32 f16) * B(32x16 f16) + C(16x16 f32)
  return __builtin_amdgcn_wmma_f32_16x16x32_f16(
      /*neg_a=*/false, a, /*neg_b=*/false, b,
      /*c_mod=*/(short)0, c, /*reuse_a=*/false, /*reuse_b=*/false);
}

// ---------------------------------------------------------------------------
// Pass 0: elementwise f32 -> f16 (8 elems/thread, b128 in / b128 out).
// ---------------------------------------------------------------------------
__global__ __launch_bounds__(256)
void cvt_f16_kernel(const float* __restrict__ src, _Float16* __restrict__ dst,
                    int n8) {
  const int i = blockIdx.x * 256 + threadIdx.x;
  if (i >= n8) return;
  *(v8h*)(dst + (size_t)i * 8) = cvt8h(src + (size_t)i * 8);
}

// ---------------------------------------------------------------------------
// Pass 1: projections.  grid = (BT/16/8, 6), block = 256 (8 waves).
// Wave w owns one 16-token tile; blockIdx.y = which*2 + column-half.
// Each wave computes 4 output tiles (64 head dims) -> 32 VGPR accumulator.
// The 64x32 f16 W slice per K-step is staged into LDS with one async b128
// load per thread, double-buffered; last K-step peeled out of the loop.
// ---------------------------------------------------------------------------
__global__ __launch_bounds__(256)
void attn_proj_kernel(const _Float16* __restrict__ xh,
                      const _Float16* __restrict__ wh,
                      _Float16* __restrict__ qf,
                      _Float16* __restrict__ kf,
                      _Float16* __restrict__ vtf) {
  __shared__ _Float16 wbuf[2][64 * 32];   // 2 x 4 KB

  const int lane  = threadIdx.x & 31;
  const int wave  = threadIdx.x >> 5;
  const int g     = lane >> 4;     // half-wave (0/1)
  const int lm    = lane & 15;
  const int row0  = (blockIdx.x * 8 + wave) * 16;  // flat token row
  const int which = blockIdx.y >> 1;               // 0=q, 1=k, 2=v
  const int nrow0 = (blockIdx.y & 1) * 64;         // head-dim half
  const _Float16* Wsel = wh + (size_t)which * HEAD * EMBED;

  v8f zero = {};
  v8f acc[4];
#pragma unroll
  for (int i = 0; i < 4; ++i) acc[i] = zero;

  const _Float16* xrow = xh + (size_t)(row0 + lm) * EMBED;

  // Stage W[nrow0:nrow0+64, c0:c0+32] into wbuf[buf]: 256 x 16B chunks.
  auto stage = [&](int c0, int buf) {
    const int i   = threadIdx.x;   // chunk id 0..255
    const int row = i >> 2;        // 0..63
    const int col = (i & 3) * 8;
    const _Float16* gp = Wsel + (size_t)(nrow0 + row) * EMBED + c0 + col;
    _Float16* lp = &wbuf[buf][row * 32 + col];
#if HAVE_ASYNC
    __builtin_amdgcn_global_load_async_to_lds_b128(
        (gptr_b128)gp, (lptr_b128)lp, /*offset=*/0, /*cpol=*/0);
#else
    *(v8h*)lp = *(const v8h*)gp;   // register relay fallback
#endif
  };

  // One K-step of WMMAs out of wbuf[cur].
  auto compute = [&](int c0, int cur) {
    v16h a = cat16(ld8h(xrow + c0 + 8 * g), ld8h(xrow + c0 + 16 + 8 * g));
#pragma unroll
    for (int nt = 0; nt < 4; ++nt) {
      const _Float16* wr = &wbuf[cur][(nt * 16 + lm) * 32];
      v16h b = cat16(*(const v8h*)(wr + 8 * g), *(const v8h*)(wr + 16 + 8 * g));
      acc[nt] = wmma16(a, b, acc[nt]);
    }
  };

  stage(0, 0);
  for (int k = 0; k < KSTEPS - 1; ++k) {     // branch-free body, single latch
    stage((k + 1) * 32, (k & 1) ^ 1);
    WAIT_ASYNC(1);   // async loads complete in order: slice k is resident
    __syncthreads();
    compute(k * 32, k & 1);
    __syncthreads();   // all waves done reading before slice k+2 overwrites
  }
  WAIT_ASYNC(0);       // peeled final K-step
  __syncthreads();
  compute((KSTEPS - 1) * 32, (KSTEPS - 1) & 1);

  const float SCALE = 0.08838834764831845f;  // HEAD^-0.5, folded into q

  if (which == 2) {
    // v stored transposed: vt[b][h][t]; lane's 8 rows are consecutive t.
    const int b  = row0 / TSEQ;
    const int tl = row0 % TSEQ;
#pragma unroll
    for (int nt = 0; nt < 4; ++nt) {
      const int h = nrow0 + nt * 16 + lm;
      v8h o;
#pragma unroll
      for (int r = 0; r < 8; ++r) o[r] = (_Float16)acc[nt][r];
      *(v8h*)(vtf + ((size_t)b * HEAD + h) * TSEQ + tl + 8 * g) = o;
    }
  } else {
    _Float16* outp = (which == 0) ? qf : kf;
    const float sc = (which == 0) ? SCALE : 1.0f;
#pragma unroll
    for (int nt = 0; nt < 4; ++nt) {
#pragma unroll
      for (int r = 0; r < 8; ++r) {
        outp[(size_t)(row0 + r + 8 * g) * HEAD + nrow0 + nt * 16 + lm] =
            (_Float16)(acc[nt][r] * sc);
      }
    }
  }
}

// ---------------------------------------------------------------------------
// Pass 2: flash attention.  grid = (T/16, B), block = 32 (one wave).
// ---------------------------------------------------------------------------
__global__ __launch_bounds__(32)
void attn_flash_kernel(const _Float16* __restrict__ qf,
                       const _Float16* __restrict__ kf,
                       const _Float16* __restrict__ vtf,
                       float* __restrict__ out) {
  __shared__ _Float16 pbuf[16 * 32];   // P tile restage: C/D -> A layout

  const int lane = threadIdx.x & 31;
  const int g    = lane >> 4;
  const int lm   = lane & 15;
  const int b    = blockIdx.y;
  const int t0   = blockIdx.x * 16;

  const _Float16* qb = qf + (size_t)b * TSEQ * HEAD;
  const _Float16* kb = kf + (size_t)b * TSEQ * HEAD;
  const _Float16* vb = vtf + (size_t)b * HEAD * TSEQ;

  // Q A-fragments: 16x128 split into 4x (16x32).
  v16h Qfr[4];
  const _Float16* qrow = qb + (size_t)(t0 + lm) * HEAD;
#pragma unroll
  for (int f = 0; f < 4; ++f)
    Qfr[f] = cat16(ld8h(qrow + 32 * f + 8 * g),
                   ld8h(qrow + 32 * f + 16 + 8 * g));

  v8f zero = {};
  v8f O[8];
#pragma unroll
  for (int i = 0; i < 8; ++i) O[i] = zero;

  const float NEG = -__builtin_inff();
  float mrow[8], lrow[8];
#pragma unroll
  for (int r = 0; r < 8; ++r) { mrow[r] = NEG; lrow[r] = 0.0f; }

  for (int j0 = 0; j0 < t0 + 16; j0 += 32) {
    // Prefetch next key block (global_prefetch_b8).
    if (j0 + 32 < t0 + 16) {
      __builtin_prefetch(kb + (size_t)(j0 + 32 + lm) * HEAD, 0, 0);
      __builtin_prefetch(kb + (size_t)(j0 + 48 + lm) * HEAD, 0, 0);
    }

    // S = Q * K^T   (two 16x16 tiles: keys j0..j0+15 and j0+16..j0+31)
    v8f S0 = zero, S1 = zero;
    const _Float16* krow0 = kb + (size_t)(j0 + lm) * HEAD;
    const _Float16* krow1 = kb + (size_t)(j0 + 16 + lm) * HEAD;
#pragma unroll
    for (int f = 0; f < 4; ++f) {
      v16h B0 = cat16(ld8h(krow0 + 32 * f + 8 * g),
                      ld8h(krow0 + 32 * f + 16 + 8 * g));
      v16h B1 = cat16(ld8h(krow1 + 32 * f + 8 * g),
                      ld8h(krow1 + 32 * f + 16 + 8 * g));
      S0 = wmma16(Qfr[f], B0, S0);
      S1 = wmma16(Qfr[f], B1, S1);
    }

    // Causal mask: C/D layout -> lane covers col lm (+16), row r + 8g.
#pragma unroll
    for (int r = 0; r < 8; ++r) {
      const int row = t0 + r + 8 * g;
      if (j0 + lm > row)      S0[r] = NEG;
      if (j0 + 16 + lm > row) S1[r] = NEG;
    }

    // Online softmax; row reductions across the 16-lane half-wave.
    float alpha[8];
#pragma unroll
    for (int r = 0; r < 8; ++r) {
      float mx = fmaxf(S0[r], S1[r]);
      mx = fmaxf(mx, __shfl_xor(mx, 1));
      mx = fmaxf(mx, __shfl_xor(mx, 2));
      mx = fmaxf(mx, __shfl_xor(mx, 4));
      mx = fmaxf(mx, __shfl_xor(mx, 8));
      const float mnew = fmaxf(mrow[r], mx);
      const float a  = __expf(mrow[r] - mnew);
      const float p0 = __expf(S0[r] - mnew);
      const float p1 = __expf(S1[r] - mnew);
      float rs = p0 + p1;
      rs += __shfl_xor(rs, 1);
      rs += __shfl_xor(rs, 2);
      rs += __shfl_xor(rs, 4);
      rs += __shfl_xor(rs, 8);
      lrow[r] = lrow[r] * a + rs;
      mrow[r] = mnew;
      alpha[r] = a;
      S0[r] = p0;
      S1[r] = p1;
    }

    // Rescale running output.
#pragma unroll
    for (int nt = 0; nt < 8; ++nt)
#pragma unroll
      for (int r = 0; r < 8; ++r) O[nt][r] *= alpha[r];

    // Restage P (C/D layout) into A-fragment layout through LDS.
#pragma unroll
    for (int r = 0; r < 8; ++r) {
      pbuf[(r + 8 * g) * 32 + lm]      = (_Float16)S0[r];
      pbuf[(r + 8 * g) * 32 + 16 + lm] = (_Float16)S1[r];
    }
    __syncthreads();   // single-wave WG: cheap, forces ds ordering
    const _Float16* pr = pbuf + lm * 32;
    v16h Pa = cat16(*(const v8h*)(pr + 8 * g), *(const v8h*)(pr + 16 + 8 * g));
    __syncthreads();

    // O += P * V   (V transposed: vt[h][t] -> contiguous b128 per fragment)
#pragma unroll
    for (int nt = 0; nt < 8; ++nt) {
      const _Float16* vrow = vb + (size_t)(nt * 16 + lm) * TSEQ + j0;
      v16h Bv = cat16(ld8h(vrow + 8 * g), ld8h(vrow + 16 + 8 * g));
      O[nt] = wmma16(Pa, Bv, O[nt]);
    }
  }

  // Normalize and write fp32 output [B,T,H].
  float inv[8];
#pragma unroll
  for (int r = 0; r < 8; ++r) inv[r] = 1.0f / lrow[r];
#pragma unroll
  for (int nt = 0; nt < 8; ++nt)
#pragma unroll
    for (int r = 0; r < 8; ++r)
      out[((size_t)b * TSEQ + t0 + r + 8 * g) * HEAD + nt * 16 + lm] =
          O[nt][r] * inv[r];
}

// ---------------------------------------------------------------------------
extern "C" void kernel_launch(void* const* d_in, const int* in_sizes, int n_in,
                              void* d_out, int out_size, void* d_ws, size_t ws_size,
                              hipStream_t stream) {
  // setup_inputs() dict order: x, Wk, Wq, Wv
  const float* x  = (const float*)d_in[0];
  const float* Wk = (const float*)d_in[1];
  const float* Wq = (const float*)d_in[2];
  const float* Wv = (const float*)d_in[3];
  float* out = (float*)d_out;

  // Workspace layout (f16): xh[BT*EMBED] | wh[3*HEAD*EMBED] | q | k | vt
  _Float16* xh  = (_Float16*)d_ws;
  _Float16* wh  = xh + (size_t)BT * EMBED;
  _Float16* qf  = wh + (size_t)3 * HEAD * EMBED;
  _Float16* kf  = qf + (size_t)BT * HEAD;
  _Float16* vtf = kf + (size_t)BT * HEAD;

  // Pass 0: one-time f32 -> f16 conversions.
  const int xn8 = BT * EMBED / 8;          // 2,097,152
  const int wn8 = HEAD * EMBED / 8;        // 16,384
  cvt_f16_kernel<<<(xn8 + 255) / 256, 256, 0, stream>>>(x, xh, xn8);
  cvt_f16_kernel<<<(wn8 + 255) / 256, 256, 0, stream>>>(Wq, wh + 0 * (size_t)HEAD * EMBED, wn8);
  cvt_f16_kernel<<<(wn8 + 255) / 256, 256, 0, stream>>>(Wk, wh + 1 * (size_t)HEAD * EMBED, wn8);
  cvt_f16_kernel<<<(wn8 + 255) / 256, 256, 0, stream>>>(Wv, wh + 2 * (size_t)HEAD * EMBED, wn8);

  // Pass 1: projections (q scaled, k, v transposed), f16 outputs.
  dim3 pgrid(BT / 16 / 8, 6);
  attn_proj_kernel<<<pgrid, 256, 0, stream>>>(xh, wh, qf, kf, vtf);

  // Pass 2: causal flash attention.
  dim3 agrid(TSEQ / 16, BATCH);
  attn_flash_kernel<<<agrid, 32, 0, stream>>>(qf, kf, vtf, out);
}
